// SmallGBNModel_31954556682582
// MI455X (gfx1250) — matrix operand order repaired and checked by reading, not compiled, not run on hardware
//
#include <hip/hip_runtime.h>
#include <math.h>

typedef __attribute__((ext_vector_type(16))) __bf16 v16bf;
typedef __attribute__((ext_vector_type(8)))  float  v8f;

#define B_TOTAL   131072
#define ROWLEN    128      // 16 channels * 8 blades
#define EPSF      1e-6f
#define WAVES_PB  8
#define NTILES    (B_TOTAL / 16)          // 8192 tiles of 16 rows
#define NBLOCKS   (NTILES / WAVES_PB)     // 1024

struct alignas(16) U4 { unsigned int x, y, z, w; };
struct alignas(32) U8 { unsigned int u[8]; };

// ---- Clifford helpers (p=3, q=0) ----
__device__ __forceinline__ float csign(int a, int b) {
  int par = 0;
  int aa = a >> 1;
  while (aa) { par ^= __popc(aa & b); aa >>= 1; }
  return (par & 1) ? -1.0f : 1.0f;
}
__device__ __forceinline__ float revsign(int k) {
  int g = __popc(k);
  return (((g * (g - 1)) / 2) & 1) ? -1.0f : 1.0f;
}
__device__ __forceinline__ unsigned short f2bf(float f) {
  unsigned int u = __float_as_uint(f);
  u += 0x7FFFu + ((u >> 16) & 1u);          // round-to-nearest-even
  return (unsigned short)(u >> 16);
}
__device__ __forceinline__ float bf2f(unsigned short h) {
  return __uint_as_float(((unsigned int)h) << 16);
}
__device__ __forceinline__ float gelu_exact(float v) {
  return 0.5f * v * (1.0f + erff(v * 0.70710678118654752f));
}

// ============================================================================
// Kernel 1: precompute per-channel 8x8 sandwich matrix M' (ln_scale folded in)
// and the 128x128 CliffordLinear matrix W2, split hi/lo bf16, stored directly
// in WMMA B-fragment order: [kb][nb][lane][v] packed bf16 pairs (dwords).
// B order used here (32x16 bf16, contiguous K, same as A): lanes 0-15 -> K=0..15, lanes 16-31 -> K=16..31;
// VGPR v holds K = half*16 + 2v, 2v+1; N = lane&15.
// ============================================================================
__global__ void gbn_prep(const float* __restrict__ ln_scale,
                         const float* __restrict__ rotor,
                         const float* __restrict__ w,
                         float* __restrict__ Mout,          // [16][8][8]
                         unsigned int* __restrict__ bh,     // 8192 dwords
                         unsigned int* __restrict__ bl) {   // 8192 dwords
  int tid = threadIdx.x;

  // M'[c][m][j] = ln_scale[c] * sum_k Bmat[m][k] * Amat[k][j]
  //   Amat[k][j] = r_hat[k^j] * s(k^j, j)
  //   Bmat[m][k] = r_hat[k^m] * rev(k^m) * s(k, k^m)
  for (int idx = tid; idx < 16 * 64; idx += 256) {
    int c = idx >> 6, m = (idx >> 3) & 7, j = idx & 7;
    float s2 = EPSF;
#pragma unroll
    for (int i = 0; i < 8; ++i) { float v = rotor[c * 8 + i]; s2 += v * v; }
    float inv = 1.0f / sqrtf(s2);
    float acc = 0.0f;
#pragma unroll
    for (int k = 0; k < 8; ++k) {
      float Bm = rotor[c * 8 + (k ^ m)] * inv * revsign(k ^ m) * csign(k, k ^ m);
      float Ak = rotor[c * 8 + (k ^ j)] * inv * csign(k ^ j, j);
      acc += Bm * Ak;
    }
    Mout[idx] = acc * ln_scale[c];
  }

  // W2[(c,j)][(o,kk)] = s(j^kk, j) * w[o][c][j^kk], pre-swizzled B fragments
  for (int idx = tid; idx < 8192; idx += 256) {
    int v = idx & 7, lane = (idx >> 3) & 31, nb = (idx >> 8) & 7, kb = idx >> 11;
    int half = lane >> 4, n = lane & 15;
    int K = kb * 32 + half * 16 + 2 * v;    // even
    int col = nb * 16 + n;                  // = o*8 + kk
    int o = col >> 3, kk = col & 7;
    int c0 = K >> 3, j0 = K & 7;
    int c1 = (K + 1) >> 3, j1 = (K + 1) & 7;
    float e0 = csign(j0 ^ kk, j0) * w[(o * 16 + c0) * 8 + (j0 ^ kk)];
    float e1 = csign(j1 ^ kk, j1) * w[(o * 16 + c1) * 8 + (j1 ^ kk)];
    unsigned short h0 = f2bf(e0), h1 = f2bf(e1);
    bh[idx] = (unsigned int)h0 | ((unsigned int)h1 << 16);
    unsigned short l0 = f2bf(e0 - bf2f(h0)), l1 = f2bf(e1 - bf2f(h1));
    bl[idx] = (unsigned int)l0 | ((unsigned int)l1 << 16);
  }
}

// ============================================================================
// Kernel 2: per wave: preprocess one 16x128 tile (LN -> 8x8 matvec -> GELU),
// split to bf16 hi/lo in LDS, then 16x128 @ 128x128 GEMM via bf16 WMMA with
// the 3-term split-precision scheme, f32 accumulation, bias at store.
// ============================================================================
__global__ __launch_bounds__(256) void gbn_main(
    const float* __restrict__ x,
    const float* __restrict__ Mg,          // [16][64]
    const float* __restrict__ bias,        // [128]
    const unsigned int* __restrict__ bhG,  // 8192 dwords
    const unsigned int* __restrict__ blG,  // 8192 dwords
    float* __restrict__ out) {
  extern __shared__ unsigned int smem[];
  unsigned int* w2h = smem;                 // 8192 dwords (32 KB)
  unsigned int* w2l = smem + 8192;          // 8192 dwords (32 KB)
  unsigned int* actB = smem + 16384;        // 8 waves * 4096 dwords (128 KB)

  const int tid = threadIdx.x;
  const int lane = tid & 31;
  const int wave = tid >> 5;
  const int half = lane >> 4;
  const int c = lane & 15;                  // this lane's fixed channel
  unsigned int* actH = actB + wave * 4096;  // [16][64] packed bf16-hi pairs
  unsigned int* actL = actH + 2048;         // [16][64] packed bf16-lo pairs

  // --- cooperative W2 fragments -> LDS (global copy stays hot in L2) ---
  for (int i = tid; i < 2048; i += 256) {
    *(U4*)&w2h[i * 4] = *(const U4*)&bhG[i * 4];
    *(U4*)&w2l[i * 4] = *(const U4*)&blG[i * 4];
  }

  // --- per-lane rotor matrix in registers ---
  float M[64];
#pragma unroll
  for (int i = 0; i < 64; ++i) M[i] = Mg[c * 64 + i];

  float bcol[8];
#pragma unroll
  for (int nb = 0; nb < 8; ++nb) bcol[nb] = bias[nb * 16 + (lane & 15)];

  const int tile = blockIdx.x * WAVES_PB + wave;      // 0..8191
  const float* xr = x + (size_t)tile * 16 * ROWLEN;

  // --- preprocessing: 8 multivectors per lane (rows 2t+half, channel c) ---
#pragma unroll
  for (int t = 0; t < 8; ++t) {
    int r = 2 * t + half;
    float4 a = *(const float4*)(xr + r * ROWLEN + c * 8);
    float4 b = *(const float4*)(xr + r * ROWLEN + c * 8 + 4);
    float xv[8] = {a.x, a.y, a.z, a.w, b.x, b.y, b.z, b.w};
    float s2 = EPSF;
#pragma unroll
    for (int j = 0; j < 8; ++j) s2 += xv[j] * xv[j];
    float inv = 1.0f / sqrtf(s2);
    float g[8];
#pragma unroll
    for (int m = 0; m < 8; ++m) {
      float y = 0.0f;
#pragma unroll
      for (int j = 0; j < 8; ++j) y += M[m * 8 + j] * xv[j];
      g[m] = gelu_exact(y * inv);
    }
#pragma unroll
    for (int d = 0; d < 4; ++d) {
      float g0 = g[2 * d], g1 = g[2 * d + 1];
      unsigned short h0 = f2bf(g0), h1 = f2bf(g1);
      actH[r * 64 + c * 4 + d] = (unsigned int)h0 | ((unsigned int)h1 << 16);
      unsigned short l0 = f2bf(g0 - bf2f(h0)), l1 = f2bf(g1 - bf2f(h1));
      actL[r * 64 + c * 4 + d] = (unsigned int)l0 | ((unsigned int)l1 << 16);
    }
  }

  __syncthreads();   // uniform: all 8 waves per block take the same path

  // --- WMMA: acc[nb] += Ah*Bh + Ah*Bl + Al*Bh, K = 4 blocks of 32 ---
  v8f acc[8];
#pragma unroll
  for (int i = 0; i < 8; ++i) acc[i] = (v8f){0, 0, 0, 0, 0, 0, 0, 0};

  const int m = lane & 15;  // A-matrix row for this lane
#pragma unroll
  for (int kb = 0; kb < 4; ++kb) {
    // A fragment (16x32 bf16): VGPR j -> K=half*16+2j (contiguous, matches B)
    U8 a8h, a8l;
    *(U4*)&a8h.u[0] = *(const U4*)&actH[m * 64 + kb * 16 + half * 8];
    *(U4*)&a8h.u[4] = *(const U4*)&actH[m * 64 + kb * 16 + 4 + half * 8];
    *(U4*)&a8l.u[0] = *(const U4*)&actL[m * 64 + kb * 16 + half * 8];
    *(U4*)&a8l.u[4] = *(const U4*)&actL[m * 64 + kb * 16 + 4 + half * 8];
    v16bf Ah = __builtin_bit_cast(v16bf, a8h);
    v16bf Al = __builtin_bit_cast(v16bf, a8l);
#pragma unroll
    for (int nb = 0; nb < 8; ++nb) {
      int bbase = ((kb * 8 + nb) * 32 + lane) * 8;
      U8 b8h, b8l;
      *(U4*)&b8h.u[0] = *(const U4*)&w2h[bbase];
      *(U4*)&b8h.u[4] = *(const U4*)&w2h[bbase + 4];
      *(U4*)&b8l.u[0] = *(const U4*)&w2l[bbase];
      *(U4*)&b8l.u[4] = *(const U4*)&w2l[bbase + 4];
      v16bf Bh = __builtin_bit_cast(v16bf, b8h);
      v16bf Bl = __builtin_bit_cast(v16bf, b8l);
      acc[nb] = __builtin_amdgcn_wmma_f32_16x16x32_bf16(
          false, Ah, false, Bh, (short)0, acc[nb], false, false);
      acc[nb] = __builtin_amdgcn_wmma_f32_16x16x32_bf16(
          false, Ah, false, Bl, (short)0, acc[nb], false, false);
      acc[nb] = __builtin_amdgcn_wmma_f32_16x16x32_bf16(
          false, Al, false, Bh, (short)0, acc[nb], false, false);
    }
  }

  // --- store: D layout: lanes 0-15 N=lane, M=vgpr; lanes 16-31 N=lane-16, M=8+vgpr
  const int n = lane & 15;
#pragma unroll
  for (int nb = 0; nb < 8; ++nb) {
#pragma unroll
    for (int r = 0; r < 8; ++r) {
      size_t grow = (size_t)tile * 16 + r + 8 * half;
      out[grow * ROWLEN + nb * 16 + n] = acc[nb][r] + bcol[nb];
    }
  }
}

extern "C" void kernel_launch(void* const* d_in, const int* in_sizes, int n_in,
                              void* d_out, int out_size, void* d_ws, size_t ws_size,
                              hipStream_t stream) {
  (void)in_sizes; (void)n_in; (void)out_size; (void)ws_size;
  const float* x        = (const float*)d_in[0];
  const float* ln_scale = (const float*)d_in[1];
  const float* rotor    = (const float*)d_in[2];
  const float* w        = (const float*)d_in[3];
  const float* bias     = (const float*)d_in[4];
  float* out = (float*)d_out;

  unsigned int* bh = (unsigned int*)d_ws;        // 32 KB
  unsigned int* bl = bh + 8192;                  // 32 KB
  float* Mp = (float*)(bl + 8192);               // 4 KB

  gbn_prep<<<1, 256, 0, stream>>>(ln_scale, rotor, w, Mp, bh, bl);
  gbn_main<<<NBLOCKS, 256, 131072 /* 128 KB dynamic LDS */, stream>>>(
      x, Mp, bias, bh, bl, out);
}
